// HGAT_35347580846252
// MI455X (gfx1250) — matrix operand order repaired and checked by reading, not compiled, not run on hardware
//
#include <hip/hip_runtime.h>

#define SLOPE 0.2f
#define BN_EPS 1e-5f

constexpr int kN   = 8192;
constexpr int kT   = 32;
constexpr int kDM  = 128;
constexpr int kM   = 1024;
constexpr int kNNZ = 131072;
constexpr int kDIN = kT * kDM; // 4096

constexpr int kLdsStride = 136; // floats; 544B rows: 16B aligned, bank-conflict-free

typedef __attribute__((ext_vector_type(16))) __bf16 v16bf;
typedef __attribute__((ext_vector_type(8)))  float  v8f;

__device__ __forceinline__ float leaky(float x) { return x > 0.f ? x : SLOPE * x; }

// atomic max on float via ordered-int trick (valid for mixed signs with -inf init)
__device__ __forceinline__ void atomicMaxF(float* addr, float val) {
  if (__float_as_int(val) >= 0)
    atomicMax((int*)addr, __float_as_int(val));
  else
    atomicMin((unsigned int*)addr, __float_as_uint(val));
}

// Stage a 32 x 128 fp32 B-slice into LDS with the CDNA5 async-DMA path
// (GLOBAL_LOAD_ASYNC_TO_LDS_B128, tracked by ASYNCcnt). 256 threads move
// 1024 16B chunks -> exactly 4 async instructions per wave per stage.
__device__ __forceinline__ void stage_b_async(const float* __restrict__ B,
                                              unsigned ldsBase, int k0, int n0,
                                              int Ndim, int t) {
#pragma unroll
  for (int it = 0; it < 4; ++it) {
    int chunk = t + 256 * it;      // 0..1023
    int r  = chunk >> 5;           // row 0..31
    int cb = chunk & 31;           // 16B chunk within the 128-float row
    const float* g = B + (size_t)(k0 + r) * Ndim + n0 + cb * 4;
    unsigned lds = ldsBase + (unsigned)(r * kLdsStride + cb * 4) * 4u;
    asm volatile("global_load_async_to_lds_b128 %0, %1, off"
                 :: "v"(lds), "v"(g) : "memory");
  }
}

// ---------------------------------------------------------------------------
// WMMA GEMM: C[M,N] = A[M,K] @ B[K,N], fp32 in/out, bf16 v_wmma compute.
// Block: 256 threads = 8 waves -> 128x128 C tile; each wave owns 16 rows x
// 128 cols (8 accumulators, A fragment reused 8x). B K-slices (32x128) are
// double-buffered in LDS via async-to-LDS DMA; A rows stream from global
// (L2-resident for every layer here).
// Fragment layouts per CDNA5 ISA 7.12.2 (wave32):
//   A (16x32 bf16): lane half h, elem j<8 -> K=8h+j ; elem 8+j -> K=16+8h+j
//   B (32x16 bf16): lane half h, elem j<8 -> K=8h+j ; elem 8+j -> K=16+8h+j, col = lane%16
//   C (16x16 f32) : vgpr v -> row = v + 8h, col = lane%16
// EPI: 0 = plain, 1 = leaky(x+bias), 2 = leaky(x+bias) + residual
// ---------------------------------------------------------------------------
template <int EPI>
__global__ void __launch_bounds__(256)
gemm_wmma(const float* __restrict__ A, const float* __restrict__ B,
          const float* __restrict__ bias, const float* __restrict__ resid,
          float* __restrict__ C, int Mdim, int Ndim, int Kdim) {
  __shared__ float sB[2][32 * kLdsStride];

  const int t    = threadIdx.x;
  const int wave = t >> 5;
  const int lane = t & 31;
  const int half = lane >> 4;
  const int l16  = lane & 15;
  const int m0 = blockIdx.y * 128 + wave * 16;
  const int n0 = blockIdx.x * 128;

  const float* arow = A + (size_t)(m0 + l16) * Kdim;

  v8f acc[8];
  const v8f vzero = {0.f, 0.f, 0.f, 0.f, 0.f, 0.f, 0.f, 0.f};
#pragma unroll
  for (int i = 0; i < 8; ++i) acc[i] = vzero;

  const unsigned ldsb[2] = {(unsigned)(size_t)&sB[0][0], (unsigned)(size_t)&sB[1][0]};

  stage_b_async(B, ldsb[0], 0, n0, Ndim, t);
  int buf = 0;

  for (int k0 = 0; k0 < Kdim; k0 += 32) {
    const bool more = (k0 + 32) < Kdim;
    if (more) {
      stage_b_async(B, ldsb[buf ^ 1], k0 + 32, n0, Ndim, t);
      // 4 next-stage ops may stay in flight; current buffer's 4 are retired
      asm volatile("s_wait_asynccnt 0x4" ::: "memory");
    } else {
      asm volatile("s_wait_asynccnt 0x0" ::: "memory");
    }
    __syncthreads();

    if (k0 + 64 < Kdim) __builtin_prefetch(arow + k0 + 64, 0, 1);

    // A fragment (direct from global, fp32 -> bf16 in registers)
    v16bf a;
#pragma unroll
    for (int j = 0; j < 8; ++j) {
      a[j]     = (__bf16)arow[k0 + 8 * half + j];
      a[8 + j] = (__bf16)arow[k0 + 16 + 8 * half + j];
    }

    const float* sb = &sB[buf][0];
#pragma unroll
    for (int nt = 0; nt < 8; ++nt) {
      v16bf b;
#pragma unroll
      for (int e = 0; e < 16; ++e)
        b[e] = (__bf16)sb[((e & 8) * 2 + 8 * half + (e & 7)) * kLdsStride + nt * 16 + l16];
      acc[nt] = __builtin_amdgcn_wmma_f32_16x16x32_bf16(false, a, false, b, (short)0,
                                                        acc[nt], false, false);
    }

    __syncthreads(); // all waves done reading before this buffer is re-staged
    buf ^= 1;
  }

#pragma unroll
  for (int nt = 0; nt < 8; ++nt) {
    const int col = n0 + nt * 16 + l16;
#pragma unroll
    for (int v = 0; v < 8; ++v) {
      int row = m0 + v + 8 * half;
      float val = acc[nt][v];
      if (EPI >= 1) val = leaky(val + bias[col]);
      if (EPI == 2) val += resid[(size_t)row * Ndim + col];
      C[(size_t)row * Ndim + col] = val;
    }
  }
}

// ---------------------------------------------------------------------------
// BatchNorm (training-mode stats) over rows, C = 128 channels.
// ---------------------------------------------------------------------------
__global__ void bn_stats(const float* __restrict__ x, float* __restrict__ sums, long rows) {
  __shared__ float ssum[256], ssq[256];
  const int t = threadIdx.x;
  const int c = t & 127;
  const int rh = t >> 7;
  const long r0 = (long)blockIdx.x * 128;
  float s = 0.f, q = 0.f;
  for (long r = r0 + rh; r < r0 + 128 && r < rows; r += 2) {
    float v = x[r * kDM + c];
    s += v; q += v * v;
  }
  ssum[t] = s; ssq[t] = q;
  __syncthreads();
  if (t < 128) {
    atomicAdd(&sums[c],       ssum[t] + ssum[t + 128]);
    atomicAdd(&sums[kDM + c], ssq[t]  + ssq[t + 128]);
  }
}

__global__ void bn_apply(float* __restrict__ x, const float* __restrict__ sums,
                         const float* __restrict__ g, const float* __restrict__ b, long rows) {
  long idx = (long)blockIdx.x * blockDim.x + threadIdx.x;
  if (idx >= rows * kDM) return;
  int c = (int)(idx & 127);
  float inv = 1.f / (float)rows;
  float m = sums[c] * inv;
  float var = sums[kDM + c] * inv - m * m;
  x[idx] = (x[idx] - m) * rsqrtf(var + BN_EPS) * g[c] + b[c];
}

__global__ void fill_f32(float* __restrict__ p, float v, long n) {
  long i = (long)blockIdx.x * blockDim.x + threadIdx.x;
  if (i < n) p[i] = v;
}

// he_attr[e] += h[n] over incidences
__global__ void scatter_rows(const float* __restrict__ h, const int* __restrict__ he_n,
                             const int* __restrict__ he_e, float* __restrict__ he_attr) {
  long idx = (long)blockIdx.x * blockDim.x + threadIdx.x;
  if (idx >= (long)kNNZ * kDM) return;
  int i = (int)(idx >> 7), c = (int)(idx & 127);
  atomicAdd(&he_attr[(size_t)he_e[i] * kDM + c], h[(size_t)he_n[i] * kDM + c]);
}

// out[r,hd] = dot(feat[r, hd*128 : hd*128+128], att[hd, attOff : attOff+128])
__global__ void row_scores(const float* __restrict__ feat, const float* __restrict__ att,
                           float* __restrict__ out, int rows, int H, int attOff) {
  int idx = blockIdx.x * blockDim.x + threadIdx.x;
  if (idx >= rows * H) return;
  int r = idx / H, hd = idx % H;
  const float* f = feat + (size_t)r * H * kDM + hd * kDM;
  const float* a = att + hd * 2 * kDM + attOff;
  float s = 0.f;
  for (int k = 0; k < kDM; ++k) s += f[k] * a[k];
  out[idx] = s;
}

__global__ void alpha_max(const float* __restrict__ an, const float* __restrict__ ae,
                          const int* __restrict__ he_n, const int* __restrict__ he_e,
                          float* __restrict__ alpha, float* __restrict__ amax, int H) {
  long idx = (long)blockIdx.x * blockDim.x + threadIdx.x;
  if (idx >= (long)kNNZ * H) return;
  int i = (int)(idx / H), hd = (int)(idx % H);
  float a = leaky(an[(size_t)he_n[i] * H + hd] + ae[(size_t)he_e[i] * H + hd]);
  alpha[idx] = a;
  atomicMaxF(&amax[(size_t)he_e[i] * H + hd], a);
}

__global__ void alpha_exp(float* __restrict__ alpha, const float* __restrict__ amax,
                          float* __restrict__ esum, const int* __restrict__ he_e, int H) {
  long idx = (long)blockIdx.x * blockDim.x + threadIdx.x;
  if (idx >= (long)kNNZ * H) return;
  int i = (int)(idx / H), hd = (int)(idx % H);
  float ex = __expf(alpha[idx] - amax[(size_t)he_e[i] * H + hd]);
  alpha[idx] = ex;
  atomicAdd(&esum[(size_t)he_e[i] * H + hd], ex);
}

__global__ void degrees(const int* __restrict__ he_n, const int* __restrict__ he_e,
                        float* __restrict__ Dn, float* __restrict__ Bm) {
  int i = blockIdx.x * blockDim.x + threadIdx.x;
  if (i >= kNNZ) return;
  atomicAdd(&Dn[he_n[i]], 1.f);
  atomicAdd(&Bm[he_e[i]], 1.f);
}

// eo[e,hd,:] += softmax_alpha * Binv[e] * xt[n,hd,:]
__global__ void scatter_eo(const float* __restrict__ alpha, const float* __restrict__ esum,
                           const float* __restrict__ Bm, const float* __restrict__ xt,
                           const int* __restrict__ he_n, const int* __restrict__ he_e,
                           float* __restrict__ eo, int H) {
  long idx = (long)blockIdx.x * blockDim.x + threadIdx.x;
  if (idx >= (long)kNNZ * H * kDM) return;
  int f = (int)(idx & 127);
  long r = idx >> 7;
  int hd = (int)(r % H);
  int i  = (int)(r / H);
  int e = he_e[i], n = he_n[i];
  float bm = Bm[e];
  float coeff = (alpha[(long)i * H + hd] / esum[(size_t)e * H + hd]) *
                (bm > 0.f ? 1.f / bm : 0.f);
  atomicAdd(&eo[((size_t)e * H + hd) * kDM + f],
            coeff * xt[((size_t)n * H + hd) * kDM + f]);
}

// no[n,hd,:] += softmax_alpha * Dinv[n] * eo[e,hd,:]
__global__ void scatter_no(const float* __restrict__ alpha, const float* __restrict__ esum,
                           const float* __restrict__ Dn, const float* __restrict__ eo,
                           const int* __restrict__ he_n, const int* __restrict__ he_e,
                           float* __restrict__ no, int H) {
  long idx = (long)blockIdx.x * blockDim.x + threadIdx.x;
  if (idx >= (long)kNNZ * H * kDM) return;
  int f = (int)(idx & 127);
  long r = idx >> 7;
  int hd = (int)(r % H);
  int i  = (int)(r / H);
  int e = he_e[i], n = he_n[i];
  float dn = Dn[n];
  float coeff = (alpha[(long)i * H + hd] / esum[(size_t)e * H + hd]) *
                (dn > 0.f ? 1.f / dn : 0.f);
  atomicAdd(&no[((size_t)n * H + hd) * kDM + f],
            coeff * eo[((size_t)e * H + hd) * kDM + f]);
}

// h += mean over heads(no) + bias   (residual connection folded in)
__global__ void combine_heads(float* __restrict__ h, const float* __restrict__ no,
                              const float* __restrict__ bh, int H) {
  long idx = (long)blockIdx.x * blockDim.x + threadIdx.x;
  if (idx >= (long)kN * kDM) return;
  long n = idx >> 7;
  int  f = (int)(idx & 127);
  float s = 0.f;
  for (int hd = 0; hd < H; ++hd) s += no[(n * H + hd) * kDM + f];
  h[idx] += s * (1.f / (float)H) + bh[f];
}

// ---------------------------------------------------------------------------
extern "C" void kernel_launch(void* const* d_in, const int* in_sizes, int n_in,
                              void* d_out, int out_size, void* d_ws, size_t ws_size,
                              hipStream_t stream) {
  const float* x    = (const float*)d_in[0];
  const int*   he_n = (const int*)d_in[1];
  const int*   he_e = (const int*)d_in[2];
  const float* W1   = (const float*)d_in[3];
  const float* b1   = (const float*)d_in[4];
  const float* g1   = (const float*)d_in[5];
  const float* be1  = (const float*)d_in[6];
  const float* Wh1  = (const float*)d_in[7];
  const float* att1 = (const float*)d_in[8];
  const float* bh1  = (const float*)d_in[9];
  const float* g2   = (const float*)d_in[10];
  const float* be2  = (const float*)d_in[11];
  const float* Wh2  = (const float*)d_in[12];
  const float* att2 = (const float*)d_in[13];
  const float* bh2  = (const float*)d_in[14];
  const float* g3   = (const float*)d_in[15];
  const float* be3  = (const float*)d_in[16];
  const float* W3   = (const float*)d_in[17];
  const float* b3   = (const float*)d_in[18];
  const float* g4   = (const float*)d_in[19];
  const float* be4  = (const float*)d_in[20];
  float* out = (float*)d_out;

  size_t off = 0;
  auto alloc = [&](size_t bytes) -> float* {
    float* p = (float*)((char*)d_ws + off);
    off += (bytes + 255) & ~(size_t)255;
    return p;
  };
  float* h      = alloc((size_t)kN * kDM * 4);
  float* heattr = alloc((size_t)kM * kDM * 4);
  float* xt     = alloc((size_t)kN * 4 * kDM * 4);
  float* et     = alloc((size_t)kM * 4 * kDM * 4);
  float* an     = alloc((size_t)kN * 4 * 4);
  float* ae     = alloc((size_t)kM * 4 * 4);
  float* amax   = alloc((size_t)kM * 4 * 4);
  float* esum   = alloc((size_t)kM * 4 * 4);
  float* alpha  = alloc((size_t)kNNZ * 4 * 4);
  float* Dn     = alloc((size_t)kN * 4);
  float* Bm     = alloc((size_t)kM * 4);
  float* eo     = alloc((size_t)kM * 4 * kDM * 4);
  float* no     = alloc((size_t)kN * 4 * kDM * 4);
  float* sums   = alloc(2 * kDM * 4);

  auto cdiv = [](long a, long b) { return (int)((a + b - 1) / b); };

  auto bn = [&](float* buf, long rows, const float* g, const float* b) {
    (void)hipMemsetAsync(sums, 0, 2 * kDM * 4, stream);
    bn_stats<<<cdiv(rows, 128), 256, 0, stream>>>(buf, sums, rows);
    bn_apply<<<cdiv(rows * kDM, 256), 256, 0, stream>>>(buf, sums, g, b, rows);
  };

  auto hconv = [&](const float* Wh, const float* att, const float* bh, int H) {
    const int Nout = H * kDM;
    // hyperedge_attr = incidence^T @ h
    (void)hipMemsetAsync(heattr, 0, (size_t)kM * kDM * 4, stream);
    scatter_rows<<<cdiv((long)kNNZ * kDM, 256), 256, 0, stream>>>(h, he_n, he_e, heattr);
    // xt = h @ Wh ; et = he_attr @ Wh   (bf16 WMMA, 128x128 block tiles)
    gemm_wmma<0><<<dim3(Nout / 128, kN / 128), 256, 0, stream>>>(h, Wh, nullptr, nullptr,
                                                                 xt, kN, Nout, kDM);
    gemm_wmma<0><<<dim3(Nout / 128, kM / 128), 256, 0, stream>>>(heattr, Wh, nullptr, nullptr,
                                                                 et, kM, Nout, kDM);
    // attention scores: alpha split into per-node + per-edge dot products
    row_scores<<<cdiv((long)kN * H, 256), 256, 0, stream>>>(xt, att, an, kN, H, 0);
    row_scores<<<cdiv((long)kM * H, 256), 256, 0, stream>>>(et, att, ae, kM, H, kDM);
    // segment softmax grouped by hyperedge
    fill_f32<<<cdiv((long)kM * H, 256), 256, 0, stream>>>(amax, -3e38f, (long)kM * H);
    alpha_max<<<cdiv((long)kNNZ * H, 256), 256, 0, stream>>>(an, ae, he_n, he_e, alpha, amax, H);
    (void)hipMemsetAsync(esum, 0, (size_t)kM * H * 4, stream);
    alpha_exp<<<cdiv((long)kNNZ * H, 256), 256, 0, stream>>>(alpha, amax, esum, he_e, H);
    // degree normalizers
    (void)hipMemsetAsync(Dn, 0, (size_t)kN * 4, stream);
    (void)hipMemsetAsync(Bm, 0, (size_t)kM * 4, stream);
    degrees<<<cdiv(kNNZ, 256), 256, 0, stream>>>(he_n, he_e, Dn, Bm);
    // node->edge then edge->node message passing
    (void)hipMemsetAsync(eo, 0, (size_t)kM * Nout * 4, stream);
    scatter_eo<<<cdiv((long)kNNZ * Nout, 256), 256, 0, stream>>>(alpha, esum, Bm, xt,
                                                                 he_n, he_e, eo, H);
    (void)hipMemsetAsync(no, 0, (size_t)kN * Nout * 4, stream);
    scatter_no<<<cdiv((long)kNNZ * Nout, 256), 256, 0, stream>>>(alpha, esum, Dn, eo,
                                                                 he_n, he_e, no, H);
    combine_heads<<<cdiv((long)kN * kDM, 256), 256, 0, stream>>>(h, no, bh, H);
  };

  // Stage 1: h = leaky_relu(x_flat @ W1 + b1); BN1
  gemm_wmma<1><<<dim3(kDM / 128, kN / 128), 256, 0, stream>>>(x, W1, b1, nullptr, h,
                                                              kN, kDM, kDIN);
  bn(h, kN, g1, be1);

  // Stage 2: hconv1 (4 heads) with residual, BN2
  hconv(Wh1, att1, bh1, 4);
  bn(h, kN, g2, be2);

  // Stage 3: hconv2 (1 head) with residual, BN3
  hconv(Wh2, att2, bh2, 1);
  bn(h, kN, g3, be3);

  // Stage 4: out = src + leaky_relu(h @ W3 + b3); BN4 over (N*T, DM)
  gemm_wmma<2><<<dim3(kDIN / 128, kN / 128), 256, 0, stream>>>(h, W3, b3, x, out,
                                                               kN, kDIN, kDM);
  bn(out, (long)kN * kT, g4, be4);
}
